// SchrodingerAttention_23862838297338
// MI455X (gfx1250) — matrix operand order repaired and checked by reading, not compiled, not run on hardware
//
#include <hip/hip_runtime.h>
#include <hip/hip_bf16.h>
#include <math.h>

// Problem constants (B, L, D from the reference)
#define Bb 4
#define Lr 4096
#define Dd 1024
#define BLr (Bb * Lr)          // 16384 rows
#define NC 8                   // scan chunks per sequence
#define CH (Lr / NC)           // 512 chunk length

// LDS B-tile: 128 rows x 32 bf16, padded to 80B row stride (bank-conflict free:
// 80*i mod 256 covers all sixteen 16B slots). 40 ushorts per row.
#define BROW 40
#define BTILE_BYTES (128 * BROW * 2)   // 10240 B per buffer

// ---------------------------------------------------------------------------
// WMMA types (CDNA5 / gfx1250, wave32)
// ---------------------------------------------------------------------------
typedef __attribute__((ext_vector_type(16))) __bf16 v16bf;
typedef __attribute__((ext_vector_type(8)))  float  v8f;
typedef __attribute__((ext_vector_type(4)))  unsigned v4u;
typedef __attribute__((ext_vector_type(8)))  unsigned v8u;

union Frag {
    uint4 q[2];   // two 16-byte loads
    v16bf v;      // 16 bf16 = 32 bytes per lane
};

// ---------------------------------------------------------------------------
// fp32 -> bf16 (round to nearest even)
// ---------------------------------------------------------------------------
__global__ __launch_bounds__(256)
void f32_to_bf16_kernel(const float* __restrict__ in,
                        unsigned short* __restrict__ out, int n) {
    int i = blockIdx.x * blockDim.x + threadIdx.x;
    if (i < n) {
        unsigned u = __float_as_uint(in[i]);
        unsigned r = u + 0x7FFFu + ((u >> 16) & 1u);
        out[i] = (unsigned short)(r >> 16);
    }
}

// ---------------------------------------------------------------------------
// TDM: DMA one 128x32 bf16 weight tile (rows [row0,row0+128) x K [k0,k0+32))
// from Wbf[Nfeat, D] into LDS at byte offset ldsOff, with 16B padding after
// every 64B row -> 80B LDS row stride. Descriptor per ISA ch.8:
//   group0: count=1 | lds_addr | global_addr[56:0] | type=2
//   group1: data_size=2B, pad_enable, pad_interval=16dw, pad_amount=4dw,
//           tensor_dim0=D, tensor_dim1=Nfeat, tile_dim0=32, tile_dim1=128,
//           tensor_dim0_stride=D
// 2D tile -> descriptor groups 2/3 omitted (VADDR2/3 = NULL).
// ---------------------------------------------------------------------------
__device__ __forceinline__ void tdm_load_b_tile(const unsigned short* Wbf,
                                                int Nfeat, int row0, int k0,
                                                unsigned ldsOff) {
    unsigned long long ga =
        (unsigned long long)(uintptr_t)(Wbf + (size_t)row0 * Dd + k0);
    v4u g0;
    g0[0] = 0x1u;                                   // count=1, user mode
    g0[1] = ldsOff;                                 // lds_addr (bytes)
    g0[2] = (unsigned)(ga & 0xffffffffu);           // global_addr[31:0]
    g0[3] = (unsigned)((ga >> 32) & 0x1ffffffu)     // global_addr[56:32]
            | (2u << 30);                           // type=2 ("image")
    v8u g1;
    g1[0] = (1u << 16)        // data_size = 1 -> 2 bytes
          | (1u << 20)        // pad_enable
          | (3u << 22)        // pad_interval code 3 -> every 16 DWORDs (64B)
          | (3u << 25);       // pad_amount code 3 -> 4 DWORDs (16B)
    g1[1] = ((unsigned)Dd & 0xffffu) << 16;             // tensor_dim0[15:0]
    g1[2] = (((unsigned)Dd >> 16) & 0xffffu)            // tensor_dim0[31:16]
          | (((unsigned)Nfeat & 0xffffu) << 16);        // tensor_dim1[15:0]
    g1[3] = (((unsigned)Nfeat >> 16) & 0xffffu)         // tensor_dim1[31:16]
          | (32u << 16);                                // tile_dim0 = 32
    g1[4] = 128u;                                       // tile_dim1 = 128
    g1[5] = (unsigned)Dd;                               // dim0_stride[31:0]
    g1[6] = 0u;                                         // dim0_stride[47:32]
    g1[7] = 0u;                                         // dim1_stride (unused, 2D)
    asm volatile("tensor_load_to_lds %0, %1" :: "s"(g0), "s"(g1) : "memory");
}

// ---------------------------------------------------------------------------
// Fused GEMM: Out[r, e] = sum_k Xbf[r,k] * Wbf[e,k]  (+ epilogue)
//   MODE 0: gate -> sigmoid   MODE 1: psi*gate -> U   MODE 2: phi raw
// Block 256 thr = 8 waves; block tile 128x128; wave tile 32x64 (2x4 WMMA
// frags, K step 32). Weight tile staged to LDS by the Tensor Data Mover,
// double buffered, overlapped with WMMA; x fragments direct from global
// (L2-resident, only 2x block-level reuse).
// ---------------------------------------------------------------------------
template <int MODE>
__global__ __launch_bounds__(256)
void gemm_wmma_kernel(const unsigned short* __restrict__ Abf,  // [BL, D]
                      const unsigned short* __restrict__ Wbf,  // [Nfeat, D]
                      const float* __restrict__ bias,          // [Nfeat]
                      const float* __restrict__ gate,          // [BL, D] or null
                      float* __restrict__ Out,                 // [BL, Nfeat]
                      int Nfeat) {
    __shared__ unsigned short Btile[2][128 * BROW];  // 2 x 10240 B, offset 0

    const int lane = threadIdx.x & 31;
    const int wid  = threadIdx.x >> 5;   // 0..7
    const int wm   = wid & 3;            // 4 waves along M
    const int wn   = wid >> 2;           // 2 waves along N
    const int R0   = blockIdx.x * 128 + wm * 32;  // wave row base
    const int C0b  = blockIdx.y * 128;            // block col base

    // A fragment (16x32 bf16): lanes 0-15 row M=lane hold K 0-7 & 16-23,
    // lanes 16-31 row M=lane-16 hold K 8-15 & 24-31 (ISA 7.12.2).
    const unsigned short* aRow =
        Abf + (size_t)(R0 + (lane & 15)) * Dd + ((lane >> 4) << 3);
    // B fragment from LDS: lane -> column e_local = 64*wn + 16*fn + (lane&15),
// lanes 0-15 hold K 0-7 & 16-23, lanes 16-31 K 8-15 & 24-31.
    const unsigned short* bBase =
        &Btile[0][0] + (size_t)(wn * 64 + (lane & 15)) * BROW + ((lane >> 4) << 3);

    const v8f vzero = {0.f, 0.f, 0.f, 0.f, 0.f, 0.f, 0.f, 0.f};
    v8f acc[2][4];
#pragma unroll
    for (int fm = 0; fm < 2; ++fm)
#pragma unroll
        for (int fn = 0; fn < 4; ++fn) acc[fm][fn] = vzero;

    // Prologue: TDM-stage buffer 0, wait, share with the workgroup.
    if (wid == 0) {
        tdm_load_b_tile(Wbf, Nfeat, C0b, 0, 0u);
        __builtin_amdgcn_s_wait_tensorcnt(0);
    }
    __syncthreads();

    int cur = 0;
    for (int k0 = 0; k0 < Dd; k0 += 32) {
        // Kick off DMA of the next K tile into the other buffer.
        if (wid == 0 && k0 + 32 < Dd) {
            tdm_load_b_tile(Wbf, Nfeat, C0b, k0 + 32,
                            (unsigned)((cur ^ 1) * BTILE_BYTES));
        }

        Frag a[2], b[4];
#pragma unroll
        for (int fm = 0; fm < 2; ++fm) {
            const unsigned short* pa = aRow + (size_t)fm * 16 * Dd + k0;
            a[fm].q[0] = *(const uint4*)(pa);
            a[fm].q[1] = *(const uint4*)(pa + 16);
        }
#pragma unroll
        for (int fn = 0; fn < 4; ++fn) {
            const unsigned short* pb =
                bBase + (size_t)cur * 128 * BROW + (size_t)fn * 16 * BROW;
            b[fn].q[0] = *(const uint4*)(pb);
            b[fn].q[1] = *(const uint4*)(pb + 16);
        }
        if (k0 + 32 < Dd)  // hint next A tile (global_prefetch_b8)
            __builtin_prefetch(aRow + k0 + 32, 0, 3);

#pragma unroll
        for (int fm = 0; fm < 2; ++fm)
#pragma unroll
            for (int fn = 0; fn < 4; ++fn)
                acc[fm][fn] = __builtin_amdgcn_wmma_f32_16x16x32_bf16(
                    false, a[fm].v, false, b[fn].v,
                    (short)0, acc[fm][fn], false, false);

        // Producer waits for the DMA it issued; everyone syncs before swap.
        if (wid == 0 && k0 + 32 < Dd)
            __builtin_amdgcn_s_wait_tensorcnt(0);
        __syncthreads();
        cur ^= 1;
    }

    // C/D layout: VGPR j -> M = j (lanes 0-15) or j+8 (lanes 16-31); N = lane&15
    const int rOff = ((lane >> 4) << 3);
#pragma unroll
    for (int fm = 0; fm < 2; ++fm) {
#pragma unroll
        for (int fn = 0; fn < 4; ++fn) {
            const int c = C0b + wn * 64 + fn * 16 + (lane & 15);
            const float bc = bias[c];
#pragma unroll
            for (int j = 0; j < 8; ++j) {
                const int r = R0 + fm * 16 + rOff + j;
                float v = acc[fm][fn][j] + bc;
                if (MODE == 0) {
                    v = 1.0f / (1.0f + __expf(-v));               // sigmoid
                } else if (MODE == 1) {
                    v *= gate[(size_t)r * Dd + (c & (Dd - 1))];   // gated psi
                }
                Out[(size_t)r * Nfeat + c] = v;
            }
        }
    }
}

// ---------------------------------------------------------------------------
// Scan pass A: per-(b, chunk, d) local recurrence  h = r*h + u, in place on U.
// r = exp(lambda*dt), lambda = (-exp(log_gamma), omega). Writes end-of-chunk
// carry. Threads ordered d-fastest so every step is a coalesced 2*D row.
// ---------------------------------------------------------------------------
__global__ __launch_bounds__(256)
void scan_chunk_kernel(float* __restrict__ U,                 // [BL, 2D] in/out
                       float* __restrict__ CR, float* __restrict__ CI,
                       const float* __restrict__ om,
                       const float* __restrict__ lgam,
                       const float* __restrict__ dtp) {
    const int t = blockIdx.x * blockDim.x + threadIdx.x;  // B*NC*D threads
    const int d = t & (Dd - 1);
    const int bc = t >> 10;
    const int c = bc & (NC - 1);
    const int b = bc >> 3;

    const float dtabs = fabsf(dtp[0]);
    const float a = __expf(lgam[d]);
    const float dec = __expf(-a * dtabs);
    float sn, cs;
    __sincosf(om[d] * dtabs, &sn, &cs);
    const float rr = dec * cs, ri = dec * sn;

    float hr = 0.f, hi = 0.f;
    float* col = U + (size_t)(b * Lr + c * CH) * (2 * Dd) + d;
    for (int l = 0; l < CH; ++l) {
        const float ur = col[0];
        const float ui = col[Dd];
        const float nr = rr * hr - ri * hi + ur;
        const float ni = rr * hi + ri * hr + ui;
        hr = nr; hi = ni;
        col[0] = hr;
        col[Dd] = hi;
        col += 2 * Dd;
    }
    CR[t] = hr;
    CI[t] = hi;
}

// ---------------------------------------------------------------------------
// Scan pass B: combine chunk carries sequentially (NC=8 steps per (b,d)),
// emitting the prefix state P entering each chunk.
// ---------------------------------------------------------------------------
__global__ __launch_bounds__(256)
void scan_carry_kernel(const float* __restrict__ CR, const float* __restrict__ CI,
                       float* __restrict__ PR, float* __restrict__ PI,
                       const float* __restrict__ om,
                       const float* __restrict__ lgam,
                       const float* __restrict__ dtp) {
    const int t = blockIdx.x * blockDim.x + threadIdx.x;  // B*D threads
    const int d = t & (Dd - 1);
    const int b = t >> 10;

    const float dtabs = fabsf(dtp[0]);
    const float a = __expf(lgam[d]);
    const float decC = __expf(-a * dtabs * (float)CH);
    float snC, csC;
    __sincosf(om[d] * dtabs * (float)CH, &snC, &csC);
    const float rcr = decC * csC, rci = decC * snC;

    float pr = 0.f, pi = 0.f;
    for (int c = 0; c < NC; ++c) {
        const int idx = (b * NC + c) * Dd + d;
        PR[idx] = pr;
        PI[idx] = pi;
        const float cr = CR[idx], ci = CI[idx];
        const float nr = rcr * pr - rci * pi + cr;
        const float ni = rcr * pi + rci * pr + ci;
        pr = nr; pi = ni;
    }
}

// ---------------------------------------------------------------------------
// Finalize: H = H_local + P * r^{t+1}  (carry fixup applied analytically),
// per-row standardization over D (biased var, eps outside sqrt), then
// out = Hn_re*phi_re + Hn_im*phi_im   ( = Re(H_proj * conj(phi)) ).
// One block per (b,l) row; 256 threads x 4 channels each.
// ---------------------------------------------------------------------------
__global__ __launch_bounds__(256)
void finalize_kernel(const float* __restrict__ HL,   // [BL, 2D] (post scan A)
                     const float* __restrict__ PHI,  // [BL, 2D]
                     const float* __restrict__ PR, const float* __restrict__ PI,
                     const float* __restrict__ om,
                     const float* __restrict__ lgam,
                     const float* __restrict__ dtp,
                     float* __restrict__ out) {
    const int r = blockIdx.x;          // 0..BL-1
    const int b = r >> 12;             // / L
    const int l = r & (Lr - 1);
    const int lloc = l & (CH - 1);
    const int c = l >> 9;              // / CH
    const float dtabs = fabsf(dtp[0]);
    const float tp = (float)(lloc + 1) * dtabs;

    float hr[4], hi[4];
    float sr = 0.f, si = 0.f, sr2 = 0.f, si2 = 0.f;
#pragma unroll
    for (int j = 0; j < 4; ++j) {
        const int d = threadIdx.x + j * 256;
        const float a = __expf(lgam[d]);
        const float er = __expf(-a * tp);
        float sn, cs;
        __sincosf(om[d] * tp, &sn, &cs);
        const float rpr = er * cs, rpi = er * sn;
        const int pidx = (b * NC + c) * Dd + d;
        const float pr = PR[pidx], pi = PI[pidx];
        const float h_r = HL[(size_t)r * 2 * Dd + d]      + pr * rpr - pi * rpi;
        const float h_i = HL[(size_t)r * 2 * Dd + Dd + d] + pr * rpi + pi * rpr;
        hr[j] = h_r; hi[j] = h_i;
        sr += h_r; si += h_i;
        sr2 += h_r * h_r; si2 += h_i * h_i;
    }

    __shared__ float red[4][256];
    red[0][threadIdx.x] = sr;  red[1][threadIdx.x] = si;
    red[2][threadIdx.x] = sr2; red[3][threadIdx.x] = si2;
    __syncthreads();
    for (int s = 128; s > 0; s >>= 1) {
        if ((int)threadIdx.x < s) {
#pragma unroll
            for (int q = 0; q < 4; ++q)
                red[q][threadIdx.x] += red[q][threadIdx.x + s];
        }
        __syncthreads();
    }
    const float inv_d = 1.0f / (float)Dd;
    const float mr = red[0][0] * inv_d, mi = red[1][0] * inv_d;
    const float vr = fmaxf(red[2][0] * inv_d - mr * mr, 0.f);
    const float vi = fmaxf(red[3][0] * inv_d - mi * mi, 0.f);
    const float ivr = 1.0f / (sqrtf(vr) + 1e-6f);
    const float ivi = 1.0f / (sqrtf(vi) + 1e-6f);

#pragma unroll
    for (int j = 0; j < 4; ++j) {
        const int d = threadIdx.x + j * 256;
        const float phr = PHI[(size_t)r * 2 * Dd + d];
        const float phj = PHI[(size_t)r * 2 * Dd + Dd + d];
        out[(size_t)r * Dd + d] =
            (hr[j] - mr) * ivr * phr + (hi[j] - mi) * ivi * phj;
    }
}

// ---------------------------------------------------------------------------
// Launch
// ---------------------------------------------------------------------------
extern "C" void kernel_launch(void* const* d_in, const int* in_sizes, int n_in,
                              void* d_out, int out_size, void* d_ws, size_t ws_size,
                              hipStream_t stream) {
    const float* x     = (const float*)d_in[0];   // [B,L,D]
    const float* omega = (const float*)d_in[1];   // [D]
    const float* lg    = (const float*)d_in[2];   // [D]
    const float* dt    = (const float*)d_in[3];   // [1]
    const float* Wpsi  = (const float*)d_in[4];   // [2D,D]
    const float* bpsi  = (const float*)d_in[5];   // [2D]
    const float* Wphi  = (const float*)d_in[6];   // [2D,D]
    const float* bphi  = (const float*)d_in[7];   // [2D]
    const float* Wg    = (const float*)d_in[8];   // [D,D]
    const float* bg    = (const float*)d_in[9];   // [D]
    float* out = (float*)d_out;

    // Workspace carve (deterministic, ~370 MB)
    char* p = (char*)d_ws;
    auto carve = [&](size_t bytes) -> char* {
        char* q = p;
        p += (bytes + 255) & ~(size_t)255;
        return q;
    };
    unsigned short* XB  = (unsigned short*)carve((size_t)BLr * Dd * 2);
    unsigned short* WPb = (unsigned short*)carve((size_t)2 * Dd * Dd * 2);
    unsigned short* WFb = (unsigned short*)carve((size_t)2 * Dd * Dd * 2);
    unsigned short* WGb = (unsigned short*)carve((size_t)Dd * Dd * 2);
    float* G   = (float*)carve((size_t)BLr * Dd * 4);
    float* U   = (float*)carve((size_t)BLr * 2 * Dd * 4);   // becomes H_local
    float* PHI = (float*)carve((size_t)BLr * 2 * Dd * 4);
    float* CR  = (float*)carve((size_t)Bb * NC * Dd * 4);
    float* CI  = (float*)carve((size_t)Bb * NC * Dd * 4);
    float* PR  = (float*)carve((size_t)Bb * NC * Dd * 4);
    float* PI  = (float*)carve((size_t)Bb * NC * Dd * 4);

    const dim3 blk(256);

    // 1) fp32 -> bf16 conversions
    f32_to_bf16_kernel<<<(BLr * Dd + 255) / 256, blk, 0, stream>>>(x, XB, BLr * Dd);
    f32_to_bf16_kernel<<<(2 * Dd * Dd + 255) / 256, blk, 0, stream>>>(Wpsi, WPb, 2 * Dd * Dd);
    f32_to_bf16_kernel<<<(2 * Dd * Dd + 255) / 256, blk, 0, stream>>>(Wphi, WFb, 2 * Dd * Dd);
    f32_to_bf16_kernel<<<(Dd * Dd + 255) / 256, blk, 0, stream>>>(Wg, WGb, Dd * Dd);

    // 2) WMMA GEMMs (TDM-staged weights): gate first (psi epilogue consumes it)
    gemm_wmma_kernel<0><<<dim3(BLr / 128, Dd / 128), blk, 0, stream>>>(
        XB, WGb, bg, nullptr, G, Dd);
    gemm_wmma_kernel<1><<<dim3(BLr / 128, (2 * Dd) / 128), blk, 0, stream>>>(
        XB, WPb, bpsi, G, U, 2 * Dd);
    gemm_wmma_kernel<2><<<dim3(BLr / 128, (2 * Dd) / 128), blk, 0, stream>>>(
        XB, WFb, bphi, nullptr, PHI, 2 * Dd);

    // 3) Chunked linear scan (exact replacement for the FFT convolution)
    scan_chunk_kernel<<<(Bb * NC * Dd) / 256, blk, 0, stream>>>(U, CR, CI, omega, lg, dt);
    scan_carry_kernel<<<(Bb * Dd) / 256, blk, 0, stream>>>(CR, CI, PR, PI, omega, lg, dt);

    // 4) Carry fixup + standardization + Re(H * conj(phi))
    finalize_kernel<<<BLr, blk, 0, stream>>>(U, PHI, PR, PI, omega, lg, dt, out);
}
